// ContractiveNodeREN_48515950576191
// MI455X (gfx1250) — matrix operand order repaired and checked by reading, not compiled, not run on hardware
//
#include <hip/hip_runtime.h>
#include <hip/hip_bf16.h>

// ContractiveNodeREN rollout for MI455X (gfx1250, wave32, WMMA).
// One wave == 16 batch rows (one WMMA M-tile) for the full T=256 rollout;
// 128 single-wave workgroups spread across the WGP array. Weights + state
// tiles live in LDS; GEMMs use v_wmma_f32_16x16x32_f16; the lower-triangular
// tanh scan uses WMMA for off-diagonal 16-blocks and a 16-lane serial solve
// for the 16x16 diagonal blocks (1 tanh per sequential step).

#define NXc     64
#define NQc     64
#define NUc     32
#define TSTEPS  256
#define BATCHc  2048
#define HSTEPc  0.05f
#define EPSc    0.01f

typedef __attribute__((ext_vector_type(16))) _Float16 v16h;
typedef __attribute__((ext_vector_type(8)))  _Float16 v8h;
typedef __attribute__((ext_vector_type(8)))  float    v8f;

// ---- workspace layout (float offsets into d_ws; total ~111 KB) ----
#define WS_P     0        // 64x64 f32
#define WS_PINV  4096     // 64x64 f32
#define WS_M2    8192     // -H2 - Chi, 64x64 f32
#define WS_Y     12288    // 64x64 f32
#define WS_LAM   16384    // 64 f32
#define WS_DIAG  16448    // D11 diagonal 16x16 blocks, [b][jp][ip], 1024 f32
#define WS_H16   17472    // packed f16 weights start here
// half offsets inside the f16 region (same packing copied to LDS):
#define H_C1   0          // [n=q][k=x]  C1[q][x]            4096
#define H_D12  4096       // [n=q][k=u]  D12[q][u]           2048
#define H_A    6144       // [n=x][k=x'] A[n][k]             4096
#define H_B1   10240      // [n=x][k=q]  B1[n][k]            4096
#define H_B2   14336      // [n=x][k=u]  B2[n][k]            2048
#define H_D11  16384      // [n=i][k=j]  D11, diag blocks 0  4096
#define H_TOT  20480

// ---------------- WMMA helpers (wave32) ----------------
__device__ __forceinline__ v8f wmma16(v16h a, v16h b, v8f c) {
  return __builtin_amdgcn_wmma_f32_16x16x32_f16(false, a, false, b, (short)0, c,
                                                false, false);
}

// A fragment (16xK f16): lane l<16 holds row l, halves K {k0..k0+7, k0+16..k0+23};
// lanes 16..31 hold same rows with K shifted by +8 (ISA 7.12.2 layout).
__device__ __forceinline__ v16h ldA(const _Float16* sm, int ld, int k0) {
  const int l = threadIdx.x;
  const _Float16* p = sm + (l & 15) * ld + k0 + ((l >> 4) << 3);
  v16h a;
  ((v8h*)&a)[0] = *(const v8h*)(p);
  ((v8h*)&a)[1] = *(const v8h*)(p + 16);
  return a;
}

// B fragment (Kx16 f16) from weight stored [N][K] row-major:
// lane l: column n = nt*16 + (l&15), K {k0+(l>>4)*8 ..+7, k0+16+(l>>4)*8 ..+7}.
__device__ __forceinline__ v16h ldB(const _Float16* sm, int nt, int k0, int ld) {
  const int l = threadIdx.x;
  const _Float16* p = sm + (nt * 16 + (l & 15)) * ld + k0 + ((l >> 4) << 3);
  v16h b;
  ((v8h*)&b)[0] = *(const v8h*)(p);
  ((v8h*)&b)[1] = *(const v8h*)(p + 16);
  return b;
}

__device__ __forceinline__ float fast_tanh(float x) {
  float ax = __builtin_fabsf(x);
  float e  = __expf(-2.0f * ax);          // v_exp_f32 path
  float t  = (1.0f - e) / (1.0f + e);
  return x < 0.0f ? -t : t;
}

// ---------------- derive kernel 1: P, lam, Y, M2, C1, D11, D12, B2 ----------------
__global__ void __launch_bounds__(256) ren_derive1_kernel(
    const float* __restrict__ Pstar, const float* __restrict__ Chi,
    const float* __restrict__ Y1, const float* __restrict__ B2,
    const float* __restrict__ D12, const float* __restrict__ X,
    float* __restrict__ ws) {
  const int tid = threadIdx.x;
  float* P    = ws + WS_P;
  float* M2   = ws + WS_M2;
  float* Y    = ws + WS_Y;
  float* lam  = ws + WS_LAM;
  float* diag = ws + WS_DIAG;
  _Float16* hb   = (_Float16*)(ws + WS_H16);
  _Float16* hC1  = hb + H_C1;
  _Float16* hD12 = hb + H_D12;
  _Float16* hB2  = hb + H_B2;
  _Float16* hD11 = hb + H_D11;

  for (int e = tid; e < 4096; e += 256) {
    const int i = e >> 6, j = e & 63;
    float s = 0.f;
    for (int k = 0; k < 64; ++k) s += Pstar[i * 64 + k] * Pstar[j * 64 + k];
    P[e] = 0.5f * s + (i == j ? EPSc : 0.f);
  }
  if (tid < 64) {  // lam = 0.5 * diag(H4)
    const float* xr = X + (64 + tid) * 128;
    float s = 0.f;
    for (int k = 0; k < 128; ++k) s += xr[k] * xr[k];
    lam[tid] = 0.5f * (s + EPSc);
  }
  __syncthreads();
  for (int e = tid; e < 4096; e += 256) {
    const int i = e >> 6, j = e & 63;
    const float* xiR = X + i * 128;
    const float* xjR = X + j * 128;
    const float* xiQ = X + (64 + i) * 128;
    const float* xjQ = X + (64 + j) * 128;
    float h1 = 0.f, h2 = 0.f, h4 = 0.f;
    for (int k = 0; k < 128; ++k) {
      h1 += xiR[k] * xjR[k];
      h2 += xiR[k] * xjQ[k];
      h4 += xiQ[k] * xjQ[k];
    }
    if (i == j) h1 += EPSc;
    Y[e]  = -0.5f * (h1 + P[e] + Y1[i * 64 + j] - Y1[j * 64 + i]);
    M2[e] = -h2 - Chi[e];
    float d = (j < i) ? (-h4 / lam[i]) : 0.f;   // D11[i][j]
    const bool dblk = ((i >> 4) == (j >> 4));
    hD11[e] = dblk ? (_Float16)0.f : (_Float16)d;  // diag blocks zeroed for WMMA
    if (dblk) {
      const int bb = i >> 4, ip = i & 15, jp = j & 15;
      diag[bb * 256 + jp * 16 + ip] = (jp < ip) ? d : 0.f;
    }
    hC1[e] = (_Float16)(Chi[j * 64 + i] / lam[i]);  // C1[q=i][x=j]
  }
  for (int e = tid; e < 2048; e += 256) {
    hD12[e] = (_Float16)D12[e];
    hB2[e]  = (_Float16)B2[e];
  }
}

// ---------------- derive kernel 2: Pinv via Gauss-Jordan (P is SPD) ----------------
__global__ void __launch_bounds__(256) ren_inv_kernel(float* __restrict__ ws) {
  __shared__ float aug[64][128];
  __shared__ float fac[64];
  __shared__ float pv;
  const int tid = threadIdx.x;
  const float* P = ws + WS_P;
  for (int e = tid; e < 8192; e += 256) {
    const int i = e >> 7, j = e & 127;
    aug[i][j] = (j < 64) ? P[i * 64 + j] : ((j - 64) == i ? 1.f : 0.f);
  }
  __syncthreads();
  for (int k = 0; k < 64; ++k) {
    if (tid == 0) pv = 1.0f / aug[k][k];
    __syncthreads();
    if (tid < 128) aug[k][tid] *= pv;
    __syncthreads();
    if (tid < 64) fac[tid] = aug[tid][k];
    __syncthreads();
    for (int e = tid; e < 8192; e += 256) {
      const int i = e >> 7, j = e & 127;
      if (i != k) aug[i][j] -= fac[i] * aug[k][j];
    }
    __syncthreads();
  }
  float* Pinv = ws + WS_PINV;
  for (int e = tid; e < 4096; e += 256) {
    const int i = e >> 6, j = e & 63;
    Pinv[e] = aug[i][64 + j];
  }
}

// ---------------- derive kernel 3: A = Pinv@Y, B1 = Pinv@M2 (to f16) ----------------
__global__ void __launch_bounds__(256) ren_derive2_kernel(float* __restrict__ ws) {
  const int tid = threadIdx.x;
  const float* Pinv = ws + WS_PINV;
  const float* M2   = ws + WS_M2;
  const float* Y    = ws + WS_Y;
  _Float16* hb  = (_Float16*)(ws + WS_H16);
  _Float16* hA  = hb + H_A;
  _Float16* hB1 = hb + H_B1;
  for (int e = tid; e < 4096; e += 256) {
    const int i = e >> 6, j = e & 63;
    float a = 0.f, b = 0.f;
    for (int k = 0; k < 64; ++k) {
      const float p = Pinv[i * 64 + k];
      a += p * Y[k * 64 + j];
      b += p * M2[k * 64 + j];
    }
    hA[e]  = (_Float16)a;
    hB1[e] = (_Float16)b;
  }
}

// ---------------- main rollout kernel: 1 wave == 16 batch rows ----------------
__global__ void __launch_bounds__(32) ren_sim_kernel(
    const float* __restrict__ xi_init, const float* __restrict__ u_log,
    const float* __restrict__ ws, float* __restrict__ out) {
  __shared__ _Float16 sWgt[H_TOT];     // C1|D12|A|B1|B2|D11   (40 KB)
  __shared__ float    sDiag[1024];     // D11 diag blocks       (4 KB)
  __shared__ _Float16 sX[16 * 64];     // xi  f16 tile          (2 KB)
  __shared__ _Float16 sU[16 * 32];     // u_t f16 tile          (1 KB)
  __shared__ _Float16 sW[16 * 64];     // w   f16 tile          (2 KB)
  __shared__ float    sPre[16 * 64];   // vbar/pre f32 tile     (4 KB)

  const int lane = threadIdx.x;        // 0..31 (wave32)
  const int half = lane >> 4;          // 0 | 1
  const int lrow = lane & 15;          // row index within tile
  const int row0 = blockIdx.x * 16;    // batch tile base

  {  // weights -> LDS; zero sW once (diag-zeroed D11 keeps stale w inert later)
    const uint4* src = (const uint4*)(ws + WS_H16);
    uint4* dst = (uint4*)sWgt;
    for (int i = lane; i < H_TOT / 8; i += 32) dst[i] = src[i];
    const uint4* s2 = (const uint4*)(ws + WS_DIAG);
    uint4* d2 = (uint4*)sDiag;
    for (int i = lane; i < 1024 / 4; i += 32) d2[i] = s2[i];
    uint4 z = {0u, 0u, 0u, 0u};
    uint4* wz = (uint4*)sW;
    for (int i = lane; i < (16 * 64) / 8; i += 32) wz[i] = z;
  }
  {  // xi_init -> sX (f16) and xi_log[:,0,:] = xi_init (half row per lane)
    const float4* xin =
        (const float4*)(xi_init + (size_t)(row0 + lrow) * NXc + half * 32);
    float4* o0 =
        (float4*)(out + (size_t)(row0 + lrow) * (TSTEPS * NXc) + half * 32);
    _Float16* xr = sX + lrow * 64 + half * 32;
#pragma unroll
    for (int q = 0; q < 8; ++q) {
      float4 v = xin[q];
      o0[q] = v;
      xr[q * 4 + 0] = (_Float16)v.x; xr[q * 4 + 1] = (_Float16)v.y;
      xr[q * 4 + 2] = (_Float16)v.z; xr[q * 4 + 3] = (_Float16)v.w;
    }
  }
  // xi carried in f32 WMMA C/D fragments: lane holds (m = r + 8*half, n = lrow)
  v8f xi_acc[4];
#pragma unroll
  for (int nt = 0; nt < 4; ++nt)
#pragma unroll
    for (int r = 0; r < 8; ++r)
      xi_acc[nt][r] =
          xi_init[(size_t)(row0 + (half << 3) + r) * NXc + nt * 16 + lrow];
  __syncthreads();

  const _Float16* sC1  = sWgt + H_C1;
  const _Float16* sD12 = sWgt + H_D12;
  const _Float16* sA   = sWgt + H_A;
  const _Float16* sB1  = sWgt + H_B1;
  const _Float16* sB2  = sWgt + H_B2;
  const _Float16* sD11 = sWgt + H_D11;

#pragma unroll 1
  for (int t = 0; t < TSTEPS; ++t) {
    {  // stage u_t (f32 -> f16), half row per lane; prefetch u_{t+1}
      const float4* up = (const float4*)(u_log +
          ((size_t)(row0 + lrow) * TSTEPS + t) * NUc + half * 16);
      _Float16* ur = sU + lrow * 32 + half * 16;
#pragma unroll
      for (int q = 0; q < 4; ++q) {
        float4 v = up[q];
        ur[q * 4 + 0] = (_Float16)v.x; ur[q * 4 + 1] = (_Float16)v.y;
        ur[q * 4 + 2] = (_Float16)v.z; ur[q * 4 + 3] = (_Float16)v.w;
      }
      if (t + 1 < TSTEPS)
        __builtin_prefetch(u_log +
            ((size_t)(row0 + lrow) * TSTEPS + t + 1) * NUc + half * 16, 0, 1);
    }
    __syncthreads();

    // vbar = xi@C1^T + u@D12^T, spilled straight to sPre
#pragma unroll
    for (int nt = 0; nt < 4; ++nt) {
      v8f acc = {};
      acc = wmma16(ldA(sX, 64, 0),  ldB(sC1, nt, 0, 64),  acc);
      acc = wmma16(ldA(sX, 64, 32), ldB(sC1, nt, 32, 64), acc);
      acc = wmma16(ldA(sU, 32, 0),  ldB(sD12, nt, 0, 32), acc);
      const int c = nt * 16 + lrow;
#pragma unroll
      for (int r = 0; r < 8; ++r) sPre[((half << 3) + r) * 64 + c] = acc[r];
    }
    __syncthreads();

    // blocked lower-triangular tanh scan over NQ=64 (4 blocks of 16)
#pragma unroll
    for (int b = 0; b < 4; ++b) {
      if (b > 0) {  // off-diagonal contribution via WMMA, RMW on sPre
        const int c = b * 16 + lrow;
        v8f pf;
#pragma unroll
        for (int r = 0; r < 8; ++r) pf[r] = sPre[((half << 3) + r) * 64 + c];
        const int nk = (16 * b + 31) >> 5;  // 1,1,2 K-chunks of 32
#pragma unroll
        for (int kc = 0; kc < nk; ++kc)
          pf = wmma16(ldA(sW, 64, kc * 32), ldB(sD11, b, kc * 32, 64), pf);
#pragma unroll
        for (int r = 0; r < 8; ++r) sPre[((half << 3) + r) * 64 + c] = pf[r];
        __syncthreads();
      }
      if (lane < 16) {  // diagonal 16x16 forward substitution; lane == row
        float pre[16];
#pragma unroll
        for (int ip = 0; ip < 16; ++ip) pre[ip] = sPre[lane * 64 + b * 16 + ip];
        const float* db = sDiag + b * 256;  // [jp][ip], lane-uniform broadcasts
        _Float16* wrow = sW + lane * 64 + b * 16;
#pragma unroll
        for (int jp = 0; jp < 16; ++jp) {
          const float wv = fast_tanh(pre[jp]);
          wrow[jp] = (_Float16)wv;
#pragma unroll
          for (int ip = jp + 1; ip < 16; ++ip)
            pre[ip] = __builtin_fmaf(wv, db[jp * 16 + ip], pre[ip]);
        }
      }
      __syncthreads();
    }

    // xi_dot = xi@A^T + w@B1^T + u@B2^T; xi += h*xi_dot; store xi_log[:,t,:]
#pragma unroll
    for (int nt = 0; nt < 4; ++nt) {
      v8f d = {};
      d = wmma16(ldA(sX, 64, 0),  ldB(sA, nt, 0, 64),   d);
      d = wmma16(ldA(sX, 64, 32), ldB(sA, nt, 32, 64),  d);
      d = wmma16(ldA(sW, 64, 0),  ldB(sB1, nt, 0, 64),  d);
      d = wmma16(ldA(sW, 64, 32), ldB(sB1, nt, 32, 64), d);
      d = wmma16(ldA(sU, 32, 0),  ldB(sB2, nt, 0, 32),  d);
      v8f xn = xi_acc[nt];
#pragma unroll
      for (int r = 0; r < 8; ++r) xn[r] = __builtin_fmaf(HSTEPc, d[r], xn[r]);
      xi_acc[nt] = xn;
      if (t > 0) {  // xi_log[:,0,:] is xi_init; states[0] is dropped
        const int c = nt * 16 + lrow;
#pragma unroll
        for (int r = 0; r < 8; ++r)
          out[(size_t)(row0 + (half << 3) + r) * (TSTEPS * NXc) +
              (size_t)t * NXc + c] = xn[r];
      }
    }
    __syncthreads();
    // publish xi_{t+1} as f16 for next step's A-operand loads
#pragma unroll
    for (int nt = 0; nt < 4; ++nt) {
      const int c = nt * 16 + lrow;
#pragma unroll
      for (int r = 0; r < 8; ++r)
        sX[((half << 3) + r) * 64 + c] = (_Float16)xi_acc[nt][r];
    }
    __syncthreads();
  }
}

extern "C" void kernel_launch(void* const* d_in, const int* in_sizes, int n_in,
                              void* d_out, int out_size, void* d_ws, size_t ws_size,
                              hipStream_t stream) {
  (void)in_sizes; (void)n_in; (void)out_size; (void)ws_size;
  const float* xi_init = (const float*)d_in[0];
  const float* u_log   = (const float*)d_in[1];
  const float* Pstar   = (const float*)d_in[2];
  const float* Chi     = (const float*)d_in[3];
  const float* Y1      = (const float*)d_in[4];
  const float* B2      = (const float*)d_in[5];
  const float* D12     = (const float*)d_in[6];
  const float* X       = (const float*)d_in[7];
  float* ws  = (float*)d_ws;   // needs ~111 KB
  float* out = (float*)d_out;

  ren_derive1_kernel<<<1, 256, 0, stream>>>(Pstar, Chi, Y1, B2, D12, X, ws);
  ren_inv_kernel<<<1, 256, 0, stream>>>(ws);
  ren_derive2_kernel<<<1, 256, 0, stream>>>(ws);
  ren_sim_kernel<<<BATCHc / 16, 32, 0, stream>>>(xi_init, u_log, ws, out);
}
